// RelativeAttention_12043088298614
// MI455X (gfx1250) — matrix operand order repaired and checked by reading, not compiled, not run on hardware
//
#include <hip/hip_runtime.h>

// Problem constants (match reference)
#define T_CTX 1024
#define NXDIM 1024
#define NHEAD 16
#define HD    64
#define BSZ   4
#define N3    3072   // 3*NX

typedef __attribute__((ext_vector_type(16))) _Float16 v16h;
typedef __attribute__((ext_vector_type(8)))  float    v8f;

// ---------------------------------------------------------------------------
// wave-local LDS ordering fence: compiler scheduling barrier + mem clobber.
// DS ops from one wave complete in order, so this is sufficient for
// store->load visibility across lanes of the same wave.
// ---------------------------------------------------------------------------
__device__ __forceinline__ void wave_lds_fence() {
  __builtin_amdgcn_wave_barrier();
  asm volatile("" ::: "memory");
}

// ---------------------------------------------------------------------------
// WMMA fragment loaders (CDNA5 wave32 layouts, CDNA5 ISA §7.12.2)
// A (16x32 f16): lane L holds row M=L&15; K elems {kb..kb+7, kb+16..kb+23},
//                kb = 8*(L>>4).  Source is row-major, k contiguous.
// B (32x16 f16): lane L holds col N=L&15; K elems {kb..kb+7, kb+16..kb+23}, kb=8*(L>>4).
//                Source is "Bt" = column-major (n-major, k contiguous).
// C/D (16x16 f32): lane L holds col N=L&15, rows (r + 8*(L>>4)) for r=0..7.
// ---------------------------------------------------------------------------
__device__ __forceinline__ v16h load_a_frag(const _Float16* src, int ld, int lane) {
  const int m  = lane & 15;
  const int kb = (lane >> 4) * 8;
  const _Float16* p = src + (size_t)m * ld + kb;
  v16h a;
#pragma unroll
  for (int i = 0; i < 8; ++i) { a[i] = p[i]; a[i + 8] = p[16 + i]; }
  return a;
}

__device__ __forceinline__ v16h load_b_frag(const _Float16* srcT, int ld, int lane) {
  const int n  = lane & 15;
  const int kb = (lane >> 4) * 8;
  const _Float16* p = srcT + (size_t)n * ld + kb;
  v16h b;
#pragma unroll
  for (int i = 0; i < 8; ++i) { b[i] = p[i]; b[i + 8] = p[16 + i]; }
  return b;
}

__device__ __forceinline__ v8f wmma_f16(v16h a, v16h b, v8f c) {
  return __builtin_amdgcn_wmma_f32_16x16x32_f16(false, a, false, b, (short)0, c,
                                                false, false);
}

// ---------------------------------------------------------------------------
// Pipelined 32x64 per-wave GEMM mainloop over K = NXDIM (f16 in, f32 acc).
// Ping-pong double buffering (2-stage unrolled body): each buffer is refilled
// only after every WMMA consuming it has issued, so the loads coalesce into
// the same registers -- no rotate copies, no WMMA->VALU WAR hazard NOPs.
// acc[0..3] = rows [0,16) x cols j*16; acc[4..7] = rows [16,32) x cols j*16.
// ---------------------------------------------------------------------------
__device__ __forceinline__ void load_slice(const _Float16* __restrict__ A0,
                                           const _Float16* __restrict__ B0,
                                           int k, int lane,
                                           v16h& a0, v16h& a1, v16h b[4]) {
  a0 = load_a_frag(A0 + k, NXDIM, lane);
  a1 = load_a_frag(A0 + (size_t)16 * NXDIM + k, NXDIM, lane);
#pragma unroll
  for (int j = 0; j < 4; ++j)
    b[j] = load_b_frag(B0 + (size_t)(j * 16) * NXDIM + k, NXDIM, lane);
}

__device__ __forceinline__ void gemm_tile_fma8(const v16h& a0, const v16h& a1,
                                               const v16h b[4], v8f acc[8]) {
#pragma unroll
  for (int j = 0; j < 4; ++j) {
    acc[j]     = wmma_f16(a0, b[j], acc[j]);
    acc[4 + j] = wmma_f16(a1, b[j], acc[4 + j]);
  }
}

__device__ __forceinline__ void gemm_mainloop_32x64(const _Float16* __restrict__ A0,
                                                    const _Float16* __restrict__ B0,
                                                    int lane, v8f acc[8]) {
  v16h a0[2], a1[2], b[2][4];
  load_slice(A0, B0, 0, lane, a0[0], a1[0], b[0]);
  for (int k = 0; k < NXDIM; k += 64) {
    // stage 0: prefetch k+32 into buffer 1, compute with buffer 0
    load_slice(A0, B0, k + 32, lane, a0[1], a1[1], b[1]);   // k+32 <= K-32 always
    gemm_tile_fma8(a0[0], a1[0], b[0], acc);
    // stage 1: prefetch k+64 into buffer 0 (except last), compute with buffer 1
    if (k + 64 < NXDIM)
      load_slice(A0, B0, k + 64, lane, a0[0], a1[0], b[0]);
    gemm_tile_fma8(a0[1], a1[1], b[1], acc);
  }
}

// ---------------------------------------------------------------------------
// Prep kernels: f32 -> f16 convert / transpose-convert (tiny, L2 resident)
// ---------------------------------------------------------------------------
__global__ void cvt_f16_kernel(const float* __restrict__ s, _Float16* __restrict__ d, int n) {
  int i = blockIdx.x * blockDim.x + threadIdx.x;
  if (i < n) d[i] = (_Float16)s[i];
}

// src is K x N row-major; dst is N x K row-major ("Bt" layout for B-fragments)
__global__ void transpose_f16_kernel(const float* __restrict__ s, _Float16* __restrict__ d,
                                     int K, int N) {
  int i = blockIdx.x * blockDim.x + threadIdx.x;
  if (i < K * N) {
    int k = i / N, n = i % N;
    d[(size_t)n * K + k] = (_Float16)s[i];
  }
}

// ---------------------------------------------------------------------------
// QKV GEMM: [4096 x 1024] @ [1024 x 3072] + b_attn.
// 8 waves/block, each wave owns a 32x64 tile (8 WMMA accumulators).
// Block tile 64 rows x 256 cols; grid (3072/256, 4096/64).
// Epilogue scatters into head-major Q (pre-scaled 1/8), K [t][d], V^T [d][t].
// ---------------------------------------------------------------------------
__global__ __launch_bounds__(256)
void qkv_gemm_kernel(const _Float16* __restrict__ xh, const _Float16* __restrict__ waT,
                     const float* __restrict__ b_attn,
                     _Float16* __restrict__ qh, _Float16* __restrict__ kh,
                     _Float16* __restrict__ vt) {
  const int lane = threadIdx.x & 31;
  const int wave = threadIdx.x >> 5;
  const int row0 = blockIdx.y * 64 + (wave >> 2) * 32;
  const int col0 = blockIdx.x * 256 + (wave & 3) * 64;

  v8f zero = {};
  v8f acc[8];
#pragma unroll
  for (int j = 0; j < 8; ++j) acc[j] = zero;

  gemm_mainloop_32x64(xh + (size_t)row0 * NXDIM, waT + (size_t)col0 * NXDIM, lane, acc);

  const int colb   = lane & 15;
  const int rowoff = (lane >> 4) * 8;
#pragma unroll
  for (int mi = 0; mi < 2; ++mi) {
#pragma unroll
    for (int j = 0; j < 4; ++j) {
#pragma unroll
      for (int r = 0; r < 8; ++r) {
        const int row = row0 + mi * 16 + rowoff + r;
        const int col = col0 + j * 16 + colb;
        const float v = acc[mi * 4 + j][r] + b_attn[col];
        const int bb = row >> 10, t = row & (T_CTX - 1);
        const int part = col >> 10, cc = col & (NXDIM - 1);
        const int h = cc >> 6, d = cc & 63;
        const size_t bh = (size_t)bb * NHEAD + h;
        if (part == 0)      qh[(bh * T_CTX + t) * HD + d] = (_Float16)(v * 0.125f);
        else if (part == 1) kh[(bh * T_CTX + t) * HD + d] = (_Float16)v;
        else                vt[(bh * HD + d) * T_CTX + t] = (_Float16)v;
      }
    }
  }
}

// ---------------------------------------------------------------------------
// Fused causal attention with relative bias + rel_values histogram trick.
// Block = 128 threads = 4 waves; each wave owns 16 query rows of one (b,h).
// Online softmax over 32-column chunks; P@V via WMMA (P staged in LDS);
// rel_values term = (normalized 64-bin histogram) @ rel_values via 2 WMMAs.
// ---------------------------------------------------------------------------
__global__ __launch_bounds__(128)
void rel_attn_kernel(const _Float16* __restrict__ qh, const _Float16* __restrict__ kh,
                     const _Float16* __restrict__ vt,
                     const int* __restrict__ rel, const float* __restrict__ relw,
                     const int* __restrict__ rel_ids, const _Float16* __restrict__ rvT,
                     _Float16* __restrict__ ah) {
  __shared__ _Float16 Pst[4][16 * 32];   // per-wave P tile (A-frag staging)
  __shared__ float    accS[4][16 * 64];  // per-wave vocab histogram (f32)
  __shared__ _Float16 accN[4][16 * 64];  // normalized histogram (A-frag staging)
  __shared__ float    frow[4][16];       // per-row rescale broadcast

  const int lane = threadIdx.x & 31;
  const int wv   = threadIdx.x >> 5;
  const int tb   = blockIdx.x & 15;      // T/64 = 16 row-blocks
  const int bh   = blockIdx.x >> 4;      // 0..63 = b*16+h
  const int b    = bh >> 4, h = bh & 15;
  const int t0   = tb * 64 + wv * 16;

  for (int i = threadIdx.x; i < 4 * 16 * 64; i += 128) ((float*)accS)[i] = 0.0f;
  __syncthreads();  // uniform: all threads reach this once

  const size_t hb = (size_t)bh * T_CTX * HD;
  const _Float16* qW = qh + hb;
  const _Float16* kW = kh + hb;
  const _Float16* vW = vt + hb;

  // Q fragments for k=0..31 and k=32..63 (held in registers for whole loop)
  const v16h qf0 = load_a_frag(qW + (size_t)t0 * HD, HD, lane);
  const v16h qf1 = load_a_frag(qW + (size_t)t0 * HD + 32, HD, lane);

  v8f zero = {};
  v8f O[4];
#pragma unroll
  for (int dt = 0; dt < 4; ++dt) O[dt] = zero;

  float m_i[8], l_i[8];
#pragma unroll
  for (int r = 0; r < 8; ++r) { m_i[r] = -1e30f; l_i[r] = 0.0f; }

  const int colb   = lane & 15;
  const int rowoff = (lane >> 4) * 8;

  const int nchunk = (t0 + 16 + 31) >> 5;   // causal: s < t0+16
  for (int ch = 0; ch < nchunk; ++ch) {
    const int s0 = ch << 5;

    // S = Q @ K^T : two 16x16 n-tiles, K-dim 64 = 2x32
    v8f sc0 = zero, sc1 = zero;
    sc0 = wmma_f16(qf0, load_b_frag(kW + (size_t)s0 * HD, HD, lane), sc0);
    sc0 = wmma_f16(qf1, load_b_frag(kW + (size_t)s0 * HD + 32, HD, lane), sc0);
    sc1 = wmma_f16(qf0, load_b_frag(kW + (size_t)(s0 + 16) * HD, HD, lane), sc1);
    sc1 = wmma_f16(qf1, load_b_frag(kW + (size_t)(s0 + 16) * HD + 32, HD, lane), sc1);

    float fs[8];
#pragma unroll
    for (int r = 0; r < 8; ++r) {
      const int t  = t0 + rowoff + r;
      const int sA = s0 + colb, sB = s0 + 16 + colb;
      float v0 = sc0[r], v1 = sc1[r];
      v0 = (sA <= t) ? v0 + relw[(size_t)rel[((size_t)b * T_CTX + t) * T_CTX + sA] * NHEAD + h]
                     : -1e30f;
      v1 = (sB <= t) ? v1 + relw[(size_t)rel[((size_t)b * T_CTX + t) * T_CTX + sB] * NHEAD + h]
                     : -1e30f;
      // row max across the 16 lanes holding this row's columns
      float mx = fmaxf(v0, v1);
#pragma unroll
      for (int msk = 8; msk >= 1; msk >>= 1) mx = fmaxf(mx, __shfl_xor(mx, msk, 32));
      const float mnew = fmaxf(m_i[r], mx);
      const float f    = __expf(m_i[r] - mnew);
      const float p0   = __expf(v0 - mnew);
      const float p1   = __expf(v1 - mnew);
      float ps = p0 + p1;
#pragma unroll
      for (int msk = 8; msk >= 1; msk >>= 1) ps += __shfl_xor(ps, msk, 32);
      l_i[r] = l_i[r] * f + ps;
      m_i[r] = mnew;
      fs[r]  = f;
      sc0[r] = p0; sc1[r] = p1;
#pragma unroll
      for (int dt = 0; dt < 4; ++dt) O[dt][r] *= f;
    }

    // broadcast per-row rescale factor, rescale histogram bins
#pragma unroll
    for (int r = 0; r < 8; ++r)
      if (colb == r) frow[wv][rowoff + r] = fs[r];
    wave_lds_fence();
    {
      const int   rr  = lane >> 1;
      const int   bb0 = (lane & 1) * 32;
      const float fr  = frow[wv][rr];
      float* ap = &accS[wv][rr * 64 + bb0];
#pragma unroll
      for (int i = 0; i < 32; ++i) ap[i] *= fr;
    }
    wave_lds_fence();

    // histogram scatter-add: hist[row][rel_ids[t,s]] += p  (ds_add_f32)
#pragma unroll
    for (int r = 0; r < 8; ++r) {
      const int t  = t0 + rowoff + r;
      const int sA = s0 + colb, sB = s0 + 16 + colb;
      if (sA <= t) atomicAdd(&accS[wv][(rowoff + r) * 64 + rel_ids[(size_t)t * T_CTX + sA]], sc0[r]);
      if (sB <= t) atomicAdd(&accS[wv][(rowoff + r) * 64 + rel_ids[(size_t)t * T_CTX + sB]], sc1[r]);
    }

    // stage P (C-layout -> row-major LDS), re-enter as A fragment, P @ V
#pragma unroll
    for (int r = 0; r < 8; ++r) {
      Pst[wv][(rowoff + r) * 32 + colb]      = (_Float16)sc0[r];
      Pst[wv][(rowoff + r) * 32 + 16 + colb] = (_Float16)sc1[r];
    }
    wave_lds_fence();
    const v16h aP = load_a_frag(&Pst[wv][0], 32, lane);
#pragma unroll
    for (int dt = 0; dt < 4; ++dt) {
      v16h bV = load_b_frag(vW + (size_t)(dt * 16) * T_CTX + s0, T_CTX, lane);
      O[dt] = wmma_f16(aP, bV, O[dt]);
    }
  }

  // finalize: normalize O, normalize histogram, add hist @ rel_values
  float inv[8];
#pragma unroll
  for (int r = 0; r < 8; ++r) {
    inv[r] = 1.0f / l_i[r];
#pragma unroll
    for (int dt = 0; dt < 4; ++dt) O[dt][r] *= inv[r];
  }
#pragma unroll
  for (int r = 0; r < 8; ++r)
    if (colb == r) frow[wv][rowoff + r] = inv[r];
  wave_lds_fence();
  {
    const int   rr  = lane >> 1;
    const int   bb0 = (lane & 1) * 32;
    const float ir  = frow[wv][rr];
#pragma unroll
    for (int i = 0; i < 32; ++i)
      accN[wv][rr * 64 + bb0 + i] = (_Float16)(accS[wv][rr * 64 + bb0 + i] * ir);
  }
  wave_lds_fence();
  {
    const v16h aR0 = load_a_frag(&accN[wv][0], 64, lane);
    const v16h aR1 = load_a_frag(&accN[wv][0] + 32, 64, lane);
#pragma unroll
    for (int dt = 0; dt < 4; ++dt) {
      O[dt] = wmma_f16(aR0, load_b_frag(rvT + (size_t)(dt * 16) * 64, 64, lane), O[dt]);
      O[dt] = wmma_f16(aR1, load_b_frag(rvT + (size_t)(dt * 16) * 64 + 32, 64, lane), O[dt]);
    }
  }

  // write a[b, t, h*64 + d] as f16 for the projection GEMM
#pragma unroll
  for (int dt = 0; dt < 4; ++dt)
#pragma unroll
    for (int r = 0; r < 8; ++r)
      ah[((size_t)b * T_CTX + t0 + rowoff + r) * NXDIM + h * HD + dt * 16 + colb] =
          (_Float16)O[dt][r];
}

// ---------------------------------------------------------------------------
// Output projection: [4096 x 1024] @ [1024 x 1024] + b_proj -> f32 out
// ---------------------------------------------------------------------------
__global__ __launch_bounds__(256)
void proj_gemm_kernel(const _Float16* __restrict__ ah, const _Float16* __restrict__ wpT,
                      const float* __restrict__ b_proj, float* __restrict__ out) {
  const int lane = threadIdx.x & 31;
  const int wave = threadIdx.x >> 5;
  const int row0 = blockIdx.y * 64 + (wave >> 2) * 32;
  const int col0 = blockIdx.x * 256 + (wave & 3) * 64;

  v8f zero = {};
  v8f acc[8];
#pragma unroll
  for (int j = 0; j < 8; ++j) acc[j] = zero;

  gemm_mainloop_32x64(ah + (size_t)row0 * NXDIM, wpT + (size_t)col0 * NXDIM, lane, acc);

  const int colb   = lane & 15;
  const int rowoff = (lane >> 4) * 8;
#pragma unroll
  for (int mi = 0; mi < 2; ++mi) {
#pragma unroll
    for (int j = 0; j < 4; ++j) {
#pragma unroll
      for (int r = 0; r < 8; ++r) {
        const int row = row0 + mi * 16 + rowoff + r;
        const int col = col0 + j * 16 + colb;
        out[(size_t)row * NXDIM + col] = acc[mi * 4 + j][r] + b_proj[col];
      }
    }
  }
}

// ---------------------------------------------------------------------------
extern "C" void kernel_launch(void* const* d_in, const int* in_sizes, int n_in,
                              void* d_out, int out_size, void* d_ws, size_t ws_size,
                              hipStream_t stream) {
  (void)in_sizes; (void)n_in; (void)out_size; (void)ws_size;
  const float* x       = (const float*)d_in[0];
  const int*   rel     = (const int*)d_in[1];
  const float* w_attn  = (const float*)d_in[2];
  const float* b_attn  = (const float*)d_in[3];
  const float* w_proj  = (const float*)d_in[4];
  const float* b_proj  = (const float*)d_in[5];
  const float* rel_w   = (const float*)d_in[6];
  const float* rel_v   = (const float*)d_in[7];
  const int*   rel_ids = (const int*)d_in[8];
  float* out = (float*)d_out;

  char* ws = (char*)d_ws;
  size_t off = 0;
  auto carve = [&](size_t bytes) -> void* {
    void* p = ws + off;
    off = (off + bytes + 255) & ~(size_t)255;
    return p;
  };

  _Float16* xh  = (_Float16*)carve((size_t)BSZ * T_CTX * NXDIM * 2);   // 8 MB
  _Float16* waT = (_Float16*)carve((size_t)N3 * NXDIM * 2);            // 6 MB
  _Float16* wpT = (_Float16*)carve((size_t)NXDIM * NXDIM * 2);         // 2 MB
  _Float16* rvT = (_Float16*)carve((size_t)64 * 64 * 2);               // 8 KB
  _Float16* qh  = (_Float16*)carve((size_t)BSZ * NHEAD * T_CTX * HD * 2); // 8 MB
  _Float16* kh  = (_Float16*)carve((size_t)BSZ * NHEAD * T_CTX * HD * 2); // 8 MB
  _Float16* vt  = (_Float16*)carve((size_t)BSZ * NHEAD * T_CTX * HD * 2); // 8 MB
  _Float16* ah  = (_Float16*)carve((size_t)BSZ * T_CTX * NXDIM * 2);   // 8 MB

  // prep: f16 conversions + weight transposes (Bt layouts)
  {
    int n = BSZ * T_CTX * NXDIM;
    cvt_f16_kernel<<<(n + 255) / 256, 256, 0, stream>>>(x, xh, n);
  }
  {
    int n = NXDIM * N3;
    transpose_f16_kernel<<<(n + 255) / 256, 256, 0, stream>>>(w_attn, waT, NXDIM, N3);
  }
  {
    int n = NXDIM * NXDIM;
    transpose_f16_kernel<<<(n + 255) / 256, 256, 0, stream>>>(w_proj, wpT, NXDIM, NXDIM);
  }
  transpose_f16_kernel<<<(64 * 64 + 255) / 256, 256, 0, stream>>>(rel_v, rvT, 64, 64);

  // qkv GEMM: block tile 64 rows x 256 cols
  qkv_gemm_kernel<<<dim3(N3 / 256, (BSZ * T_CTX) / 64), 256, 0, stream>>>(
      xh, waT, b_attn, qh, kh, vt);

  // fused relative attention
  rel_attn_kernel<<<BSZ * NHEAD * (T_CTX / 64), 128, 0, stream>>>(
      qh, kh, vt, rel, rel_w, rel_ids, rvT, ah);

  // output projection
  proj_gemm_kernel<<<dim3(NXDIM / 256, (BSZ * T_CTX) / 64), 256, 0, stream>>>(
      ah, wpT, b_proj, out);
}
